// BigramLanguageModel_56264071578075
// MI455X (gfx1250) — matrix operand order repaired and checked by reading, not compiled, not run on hardware
//
#include <hip/hip_runtime.h>
#include <hip/hip_bf16.h>

// ---------------------------------------------------------------------------
// GPT forward pass for gfx1250 (MI455X).  f16 WMMA GEMMs with f32 accumulation
// (v_wmma_f32_16x16x32_f16).  Weights converted+transposed to f16 once per
// launch; they stay resident in the 192MB L2.  One wave per (16*MT)x64 output
// tile (register blocking: A frags reused across N-tiles, B frags across
// M-tiles -> ~0.5KB of L2 traffic per WMMA).  Logits stores are non-temporal
// so the 512MB output stream does not evict L2-resident weights.
// ---------------------------------------------------------------------------

typedef __attribute__((ext_vector_type(16))) _Float16 v16h;
typedef __attribute__((ext_vector_type(8)))  _Float16 v8h;
typedef __attribute__((ext_vector_type(8)))  float    v8f;

#define Bv   8
#define Tv   500
#define Nv   512
#define Hv   8
#define HDv  64
#define Lv   6
#define Vv   32000
#define Mv   (Bv*Tv)          // 4000 rows in the token stream
#define TPAD 512              // padded score row length

static constexpr int NT = 4;  // 16-wide N-tiles per wave (64 columns)

// ----------------------------- device helpers ------------------------------

__device__ __forceinline__ float wave_sum(float v) {
#pragma unroll
  for (int m = 16; m >= 1; m >>= 1) v += __shfl_xor(v, m, 32);
  return v;
}
__device__ __forceinline__ float wave_max(float v) {
#pragma unroll
  for (int m = 16; m >= 1; m >>= 1) v = fmaxf(v, __shfl_xor(v, m, 32));
  return v;
}

// ------------------------------ WMMA GEMM ----------------------------------
// C[M,Nc] = A[M,K] (f16 row-major, lda) x B[K,Nc] given as Bt[Nc,K] (f16, ldb)
// Register blocking: MT 16-row M-tiles x NT 16-col N-tiles per wave.
// Optional per-column bias, fp32 residual add, exact GELU, f16/f32 store,
// non-temporal (streaming) f32 store.  Batched via blockIdx.z with
// (b,h) = (z/Hdiv, z%Hdiv) stride decomposition.
//
// Fragment layouts used here (wave32; A/B use this code's contiguous K order, same for both):
//  A 16x32: lane r=lane&15 is row; lanes<16 hold K 0..15,
//           lanes>=16 hold K 16..31          -> two contiguous v8h loads.
//  B 32x16: lane c=lane&15 is column; lanes<16 hold K 0..15,
//           lanes>=16 hold K 16..31          -> one contiguous v16h load.
//  C/D 16x16 f32: VGPR g holds row g (lanes 0..15) / row 8+g (lanes 16..31),
//           column = lane&15.

template<int MT, bool OUT16, bool BIAS, bool RES, bool GELU_ACT, bool STREAM>
__global__ __launch_bounds__(32)
void gemm_wmma_kernel(const _Float16* __restrict__ A,
                      const _Float16* __restrict__ Bt,
                      void* __restrict__ Cout,
                      const float* __restrict__ bias,
                      const float* __restrict__ res,
                      int M, int Nc, int K,
                      int lda, int ldb, int ldc,
                      long sAb, long sAh, long sBb, long sBh,
                      long sCb, long sCh, int Hdiv)
{
  const int lane = threadIdx.x;
  const int z  = blockIdx.z;
  const int bi = z / Hdiv;
  const int hi = z - bi * Hdiv;
  A  += (long)bi * sAb + (long)hi * sAh;
  Bt += (long)bi * sBb + (long)hi * sBh;
  const long cbase = (long)bi * sCb + (long)hi * sCh;

  const int m0 = blockIdx.x * (16 * MT);
  const int n0 = blockIdx.y * (16 * NT);
  const int r  = lane & 15;
  const int hh = lane >> 4;

  const _Float16* aptr[MT];
#pragma unroll
  for (int i = 0; i < MT; ++i) {
    int arow = m0 + i * 16 + r;
    if (arow >= M) arow = M - 1;                            // clamp (store-guarded)
    aptr[i] = A + (long)arow * lda + hh * 16;
  }

  const _Float16* bptr[NT];
#pragma unroll
  for (int j = 0; j < NT; ++j) {
    int nc = n0 + j * 16 + r;
    if (nc >= Nc) nc = Nc - 1;                              // clamp (store-guarded)
    bptr[j] = Bt + (long)nc * ldb + hh * 16;
  }

  v8f acc[MT][NT] = {};

  for (int k = 0; k < K; k += 32) {
    union { v16h v; v8h h[2]; } af[MT];
#pragma unroll
    for (int i = 0; i < MT; ++i) {
      af[i].h[0] = *(const v8h*)(aptr[i] + k);              // K = khalf .. khalf+7
      af[i].h[1] = *(const v8h*)(aptr[i] + k + 8);          // K = khalf+8 .. khalf+15
    }
    __builtin_prefetch(aptr[0] + k + 32, 0, 1);             // global_prefetch_b8
#pragma unroll
    for (int j = 0; j < NT; ++j) {
      v16h bf = *(const v16h*)(bptr[j] + k);                // 16 consecutive K
#pragma unroll
      for (int i = 0; i < MT; ++i)
        acc[i][j] = __builtin_amdgcn_wmma_f32_16x16x32_f16(
            false, af[i].v, false, bf, (short)0, acc[i][j], false, false);
    }
  }

#pragma unroll
  for (int j = 0; j < NT; ++j) {
    const int nc = n0 + j * 16 + r;
    if (nc >= Nc) continue;
    const float bv = BIAS ? bias[nc] : 0.0f;
#pragma unroll
    for (int i = 0; i < MT; ++i) {
#pragma unroll
      for (int g = 0; g < 8; ++g) {
        const int m = m0 + i * 16 + hh * 8 + g;
        if (m >= M) continue;
        float v = acc[i][j][g] + bv;
        if (RES)      v += res[cbase + (long)m * ldc + nc];
        if (GELU_ACT) v  = 0.5f * v * (1.0f + erff(v * 0.70710678118654752f));
        const long cidx = cbase + (long)m * ldc + nc;
        if (OUT16)       ((_Float16*)Cout)[cidx] = (_Float16)v;
        else if (STREAM) __builtin_nontemporal_store(v, &((float*)Cout)[cidx]);
        else             ((float*)Cout)[cidx] = v;
      }
    }
  }
}

// --------------------------- elementwise kernels ---------------------------

__global__ void embed_kernel(const int* __restrict__ idx,
                             const float* __restrict__ tok,
                             const float* __restrict__ pos,
                             float* __restrict__ x)
{
  int i = blockIdx.x * blockDim.x + threadIdx.x;
  if (i >= Mv * Nv) return;
  int mrow = i / Nv, n = i - mrow * Nv;
  int t = mrow % Tv;
  x[i] = tok[(long)idx[mrow] * Nv + n] + pos[(long)t * Nv + n];
}

// src fp32 [layer][K][Nc]  ->  dst f16 [layer][Nc][K]   (W^T for B-fragments)
__global__ void wt_transpose_kernel(const float* __restrict__ src,
                                    _Float16* __restrict__ dst, int K, int Nc)
{
  long loff = (long)blockIdx.y * K * Nc;
  int i = blockIdx.x * blockDim.x + threadIdx.x;
  if (i >= K * Nc) return;
  int k = i / Nc, n = i - k * Nc;
  dst[loff + (long)n * K + k] = (_Float16)src[loff + i];
}

// one wave per row; N=512 fixed (16 elements per lane)
__global__ __launch_bounds__(32)
void layernorm_kernel(const float* __restrict__ x,
                      const float* __restrict__ g,
                      const float* __restrict__ b,
                      _Float16* __restrict__ out)
{
  const int row = blockIdx.x, lane = threadIdx.x;
  const float* xr = x + (long)row * Nv;
  float v[16]; float s = 0.0f;
#pragma unroll
  for (int i = 0; i < 16; ++i) { v[i] = xr[lane + 32 * i]; s += v[i]; }
  s = wave_sum(s);
  const float mu = s * (1.0f / Nv);
  float q = 0.0f;
#pragma unroll
  for (int i = 0; i < 16; ++i) { float d = v[i] - mu; q += d * d; }
  q = wave_sum(q);
  const float inv = rsqrtf(q * (1.0f / Nv) + 1e-5f);
  _Float16* o = out + (long)row * Nv;
#pragma unroll
  for (int i = 0; i < 16; ++i) {
    int c = lane + 32 * i;
    o[c] = (_Float16)((v[i] - mu) * inv * g[c] + b[c]);
  }
}

// causal softmax in-place on f16 score rows [B,H,T,TPAD]; zeros the s>t tail
__global__ __launch_bounds__(32)
void softmax_kernel(_Float16* __restrict__ w, float scale)
{
  const int lane = threadIdx.x;
  const long rowIdx = blockIdx.x;                 // b*H*T + h*T + t
  const int t = (int)(rowIdx % Tv);
  _Float16* row = w + rowIdx * TPAD;
  float v[16]; float mx = -3.4e38f;
#pragma unroll
  for (int i = 0; i < 16; ++i) {
    int c = lane + 32 * i;
    float val = (c <= t) ? (float)row[c] * scale : -3.4e38f;
    v[i] = val; mx = fmaxf(mx, val);
  }
  mx = wave_max(mx);
  float s = 0.0f;
#pragma unroll
  for (int i = 0; i < 16; ++i) {
    float e = (v[i] > -1.0e37f) ? __expf(v[i] - mx) : 0.0f;
    v[i] = e; s += e;
  }
  s = wave_sum(s);
  const float invs = 1.0f / s;
#pragma unroll
  for (int i = 0; i < 16; ++i)
    row[lane + 32 * i] = (_Float16)(v[i] * invs);
}

// v f16 [B,T,N] -> vT f16 [B,H,HD,TPAD]  (zero pad t in [Tv,TPAD))
__global__ void vtrans_kernel(const _Float16* __restrict__ v,
                              _Float16* __restrict__ vT)
{
  int i = blockIdx.x * blockDim.x + threadIdx.x;
  if (i >= Bv * Hv * HDv * TPAD) return;
  int t = i & (TPAD - 1); int rest = i >> 9;
  int d = rest & (HDv - 1); rest >>= 6;
  int h = rest & (Hv - 1);  int b = rest >> 3;
  _Float16 val = (_Float16)0.0f;
  if (t < Tv) val = v[((long)(b * Tv + t)) * Nv + h * HDv + d];
  vT[i] = val;
}

// ------------------------------- launcher ----------------------------------

extern "C" void kernel_launch(void* const* d_in, const int* in_sizes, int n_in,
                              void* d_out, int out_size, void* d_ws, size_t ws_size,
                              hipStream_t stream)
{
  (void)in_sizes; (void)n_in; (void)out_size; (void)ws_size;

  const int*   idx  = (const int*)  d_in[0];
  const float* tok  = (const float*)d_in[1];
  const float* pos  = (const float*)d_in[2];
  const float* Wq   = (const float*)d_in[3];
  const float* Wk   = (const float*)d_in[4];
  const float* Wv   = (const float*)d_in[5];
  const float* Wo   = (const float*)d_in[6];
  const float* bo   = (const float*)d_in[7];
  const float* ln1g = (const float*)d_in[8];
  const float* ln1b = (const float*)d_in[9];
  const float* ln2g = (const float*)d_in[10];
  const float* ln2b = (const float*)d_in[11];
  const float* W1   = (const float*)d_in[12];
  const float* b1   = (const float*)d_in[13];
  const float* W2   = (const float*)d_in[14];
  const float* b2   = (const float*)d_in[15];
  const float* lnfg = (const float*)d_in[16];
  const float* lnfb = (const float*)d_in[17];
  const float* Wh   = (const float*)d_in[18];
  const float* bh   = (const float*)d_in[19];
  float* logits = (float*)d_out;

  // ---- workspace carve-out (256B aligned) ----
  char* p = (char*)d_ws;
  auto alloc = [&](size_t bytes) -> void* {
    void* r = (void*)p;
    p += (bytes + 255) & ~(size_t)255;
    return r;
  };
  _Float16* WqT = (_Float16*)alloc((size_t)Lv * Nv * Nv * 2);
  _Float16* WkT = (_Float16*)alloc((size_t)Lv * Nv * Nv * 2);
  _Float16* WvT = (_Float16*)alloc((size_t)Lv * Nv * Nv * 2);
  _Float16* WoT = (_Float16*)alloc((size_t)Lv * Nv * Nv * 2);
  _Float16* W1T = (_Float16*)alloc((size_t)Lv * 4 * Nv * Nv * 2);   // [l][2048][512]
  _Float16* W2T = (_Float16*)alloc((size_t)Lv * 4 * Nv * Nv * 2);   // [l][512][2048]
  _Float16* WhT = (_Float16*)alloc((size_t)Vv * Nv * 2);            // [32000][512]
  float*    x   = (float*)   alloc((size_t)Mv * Nv * 4);
  _Float16* h   = (_Float16*)alloc((size_t)Mv * Nv * 2);
  _Float16* q   = (_Float16*)alloc((size_t)Mv * Nv * 2);
  _Float16* kk  = (_Float16*)alloc((size_t)Mv * Nv * 2);
  _Float16* vv  = (_Float16*)alloc((size_t)Mv * Nv * 2);
  _Float16* vT  = (_Float16*)alloc((size_t)Bv * Hv * HDv * TPAD * 2);
  _Float16* w   = (_Float16*)alloc((size_t)Bv * Hv * Tv * TPAD * 2);
  _Float16* a   = (_Float16*)alloc((size_t)Mv * Nv * 2);
  _Float16* h1  = (_Float16*)alloc((size_t)Mv * 4 * Nv * 2);

  // ---- weight convert + transpose (L2-resident f16 B^T matrices) ----
  {
    dim3 gNN((Nv * Nv + 255) / 256, Lv);
    wt_transpose_kernel<<<gNN, 256, 0, stream>>>(Wq, WqT, Nv, Nv);
    wt_transpose_kernel<<<gNN, 256, 0, stream>>>(Wk, WkT, Nv, Nv);
    wt_transpose_kernel<<<gNN, 256, 0, stream>>>(Wv, WvT, Nv, Nv);
    wt_transpose_kernel<<<gNN, 256, 0, stream>>>(Wo, WoT, Nv, Nv);
    dim3 gMLP((Nv * 4 * Nv + 255) / 256, Lv);
    wt_transpose_kernel<<<gMLP, 256, 0, stream>>>(W1, W1T, Nv, 4 * Nv);
    wt_transpose_kernel<<<gMLP, 256, 0, stream>>>(W2, W2T, 4 * Nv, Nv);
    wt_transpose_kernel<<<dim3((Nv * Vv + 255) / 256, 1), 256, 0, stream>>>(Wh, WhT, Nv, Vv);
  }

  // ---- embeddings ----
  embed_kernel<<<(Mv * Nv + 255) / 256, 256, 0, stream>>>(idx, tok, pos, x);

  const float scale = 0.04419417382415922f;   // N^-0.5 = 1/sqrt(512)
  const int mt4 = (Mv + 63) / 64;             // 63 blocks of 64 rows (M=4000)
  const int tt2 = (Tv + 31) / 32;             // 16 blocks of 32 rows (M=500)

  for (int l = 0; l < Lv; ++l) {
    const _Float16* WqTl = WqT + (size_t)l * Nv * Nv;
    const _Float16* WkTl = WkT + (size_t)l * Nv * Nv;
    const _Float16* WvTl = WvT + (size_t)l * Nv * Nv;
    const _Float16* WoTl = WoT + (size_t)l * Nv * Nv;
    const _Float16* W1Tl = W1T + (size_t)l * 4 * Nv * Nv;
    const _Float16* W2Tl = W2T + (size_t)l * 4 * Nv * Nv;

    // LN1 -> h (f16)
    layernorm_kernel<<<Mv, 32, 0, stream>>>(x, ln1g + l * Nv, ln1b + l * Nv, h);

    // q,k,v = h @ W{q,k,v}   (f16 out, no bias)
    gemm_wmma_kernel<4,true,false,false,false,false><<<dim3(mt4, Nv/64, 1), 32, 0, stream>>>(
        h, WqTl, q, nullptr, nullptr, Mv, Nv, Nv, Nv, Nv, Nv, 0,0,0,0,0,0, 1);
    gemm_wmma_kernel<4,true,false,false,false,false><<<dim3(mt4, Nv/64, 1), 32, 0, stream>>>(
        h, WkTl, kk, nullptr, nullptr, Mv, Nv, Nv, Nv, Nv, Nv, 0,0,0,0,0,0, 1);
    gemm_wmma_kernel<4,true,false,false,false,false><<<dim3(mt4, Nv/64, 1), 32, 0, stream>>>(
        h, WvTl, vv, nullptr, nullptr, Mv, Nv, Nv, Nv, Nv, Nv, 0,0,0,0,0,0, 1);

    // vT[b,h,d,s] (padded)
    vtrans_kernel<<<(Bv*Hv*HDv*TPAD + 255) / 256, 256, 0, stream>>>(vv, vT);

    // scores[b,h,t,s] = q_bh @ k_bh^T   (Bt == k in natural layout)
    gemm_wmma_kernel<2,true,false,false,false,false><<<dim3(tt2, TPAD/64, Bv*Hv), 32, 0, stream>>>(
        q, kk, w, nullptr, nullptr, Tv, Tv, HDv, Nv, Nv, TPAD,
        (long)Tv*Nv, HDv, (long)Tv*Nv, HDv, (long)Hv*Tv*TPAD, (long)Tv*TPAD, Hv);

    // causal softmax (in place, zero-pads tail)
    softmax_kernel<<<Bv*Hv*Tv, 32, 0, stream>>>(w, scale);

    // a[b,t,h*64+d] = w_bh @ v_bh      (Bt == vT)
    gemm_wmma_kernel<2,true,false,false,false,false><<<dim3(tt2, 1, Bv*Hv), 32, 0, stream>>>(
        w, vT, a, nullptr, nullptr, Tv, HDv, TPAD, TPAD, TPAD, Nv,
        (long)Hv*Tv*TPAD, (long)Tv*TPAD, (long)Hv*HDv*TPAD, (long)HDv*TPAD,
        (long)Tv*Nv, HDv, Hv);

    // x = x + a @ Wo + bo   (f32 out, bias + residual, in place)
    gemm_wmma_kernel<4,false,true,true,false,false><<<dim3(mt4, Nv/64, 1), 32, 0, stream>>>(
        a, WoTl, x, bo + l * Nv, x, Mv, Nv, Nv, Nv, Nv, Nv, 0,0,0,0,0,0, 1);

    // LN2 -> h
    layernorm_kernel<<<Mv, 32, 0, stream>>>(x, ln2g + l * Nv, ln2b + l * Nv, h);

    // h1 = gelu(h @ W1 + b1)   (f16 out)
    gemm_wmma_kernel<4,true,true,false,true,false><<<dim3(mt4, (4*Nv)/64, 1), 32, 0, stream>>>(
        h, W1Tl, h1, b1 + l * 4 * Nv, nullptr, Mv, 4*Nv, Nv, Nv, Nv, 4*Nv,
        0,0,0,0,0,0, 1);

    // x = x + h1 @ W2 + b2   (f32 out, bias + residual, in place)
    gemm_wmma_kernel<4,false,true,true,false,false><<<dim3(mt4, Nv/64, 1), 32, 0, stream>>>(
        h1, W2Tl, x, b2 + l * Nv, x, Mv, Nv, 4*Nv, 4*Nv, 4*Nv, Nv,
        0,0,0,0,0,0, 1);
  }

  // final LN -> h
  layernorm_kernel<<<Mv, 32, 0, stream>>>(x, lnfg, lnfb, h);

  // logits = h @ Wh + bh   (f32 non-temporal stores to d_out; keep weights in L2)
  gemm_wmma_kernel<4,false,true,false,false,true><<<dim3(mt4, Vv/64, 1), 32, 0, stream>>>(
      h, WhT, logits, bh, nullptr, Mv, Vv, Nv, Nv, Nv, Vv, 0,0,0,0,0,0, 1);
}
